// Zamba2MambaMixer_87771951661353
// MI455X (gfx1250) — matrix operand order repaired and checked by reading, not compiled, not run on hardware
//
#include <hip/hip_runtime.h>
#include <hip/hip_bf16.h>
#include <cstdint>
#include <cstddef>

// ---- problem constants (match reference) ----
#define B_   2
#define L_   2048
#define DM_  2048
#define I_   4096
#define N_   64
#define H_   64
#define P_   64
#define K_   4
#define CONV_DIM_ 4224
#define PROJ_ 8384
#define EPS_ 1e-5f

// chunked-scan parameters
#define QC_  128              // chunk length (sequential steps per block)
#define NC_  (L_ / QC_)       // 16 chunks
#define CH_  8                // timesteps staged per LDS refill

typedef __attribute__((ext_vector_type(16))) __bf16 v16bf;
typedef __attribute__((ext_vector_type(8)))  float  v8f;
typedef __attribute__((ext_vector_type(4)))  unsigned int v4u;
typedef __attribute__((ext_vector_type(8)))  int v8i;
typedef __attribute__((ext_vector_type(4)))  int v4i;

#if defined(__has_builtin)
#if __has_builtin(__builtin_amdgcn_tensor_load_to_lds)
#define USE_TDM 1
#endif
#endif

static __device__ __forceinline__ unsigned short f2bf(float f) {
  union { float f; uint32_t u; } v; v.f = f;
  uint32_t u = v.u;
  u += 0x7FFFu + ((u >> 16) & 1u);   // round to nearest even
  return (unsigned short)(u >> 16);
}
static __device__ __forceinline__ float bf2f(unsigned short h) {
  union { uint32_t u; float f; } v; v.u = ((uint32_t)h) << 16;
  return v.f;
}
static __device__ __forceinline__ float siluf(float x) {
  return x / (1.0f + __expf(-x));
}

#ifdef USE_TDM
// ---------------------------------------------------------------------------
// TDM: load a 2D bf16 tile [tile_h rows x tile_w elems] from global (row
// stride = stride_elems) into LDS at byte offset lds_off, with hardware LDS
// padding of 4 DWORDs (8 ushorts) after every 16 DWORDs (32 ushorts = 1 row)
// => padded LDS row stride of 40 ushorts. Rows beyond tensor_h read as zero.
// ---------------------------------------------------------------------------
static __device__ __forceinline__ void tdm_load_tile_bf16(
    const unsigned short* gaddr, uint32_t lds_off, uint32_t tensor_h,
    uint32_t stride_elems, uint32_t tile_w, uint32_t tile_h) {
  uint64_t ga = (uint64_t)(uintptr_t)gaddr;
  v4u g0;
  g0[0] = 0x1u;                               // count=1, is_restore=0
  g0[1] = lds_off;                            // lds_addr (bytes)
  g0[2] = (uint32_t)ga;                       // global_addr[31:0]
  g0[3] = (uint32_t)(ga >> 32) | (2u << 30);  // global_addr[56:32] | type=2

  // group1: data_size=2B (code 1), pad_enable, pad_interval=16 DW (code 3),
  // pad_amount=4 DW (code 3)
  uint32_t w0 = (1u << 16) | (1u << 20) | (3u << 22) | (3u << 25);
  v8i g1;
  g1[0] = (int)w0;
  g1[1] = (int)((tile_w & 0xFFFFu) << 16);                    // tensor_dim0 lo
  g1[2] = (int)((tile_w >> 16) | ((tensor_h & 0xFFFFu) << 16)); // dim0 hi|dim1 lo
  g1[3] = (int)((tensor_h >> 16) | ((tile_w & 0xFFFFu) << 16)); // dim1 hi|tile_dim0
  g1[4] = (int)(tile_h & 0xFFFFu);                            // tile_dim1, tile_dim2=0
  g1[5] = (int)stride_elems;                                  // dim0_stride[31:0]
  g1[6] = 0;                                                  // stride hi / dim1_stride
  g1[7] = 0;

  v4i z4 = {0, 0, 0, 0};
#if __clang_major__ >= 23
  v8i z8 = {0, 0, 0, 0, 0, 0, 0, 0};
  __builtin_amdgcn_tensor_load_to_lds(g0, g1, z4, z4, z8, 0);
#else
  __builtin_amdgcn_tensor_load_to_lds(g0, g1, z4, z4, 0);
#endif
}

#if __has_builtin(__builtin_amdgcn_s_wait_tensorcnt)
#define WAIT_TENSORCNT(imm) __builtin_amdgcn_s_wait_tensorcnt(imm)
#else
#define WAIT_TENSORCNT(imm) \
  asm volatile("s_wait_tensorcnt " #imm ::: "memory")
#endif
#endif  // USE_TDM

// ---------------------------------------------------------------------------
// fp32 -> bf16 (stored as ushort) conversion
// ---------------------------------------------------------------------------
__global__ void k_cvt_bf16(const float* __restrict__ src,
                           unsigned short* __restrict__ dst, int n) {
  int i = blockIdx.x * blockDim.x + threadIdx.x;
  if (i < n) dst[i] = f2bf(src[i]);
}

// ---------------------------------------------------------------------------
// bf16 WMMA GEMM:  C[M,N] (f32) = A[M,K] (bf16, row-major) * B[N,K]^T (bf16)
// Workgroup tile 128(M) x 256(N), 8 waves (2x4), each wave 64x64 (4x4 WMMA
// tiles of 16x16), K staged in 32-element steps through double-buffered LDS.
// Tiles staged by the Tensor Data Mover (one wave issues 2 TDM descriptors
// per stage; TENSORcnt-pipelined), so no VGPRs are burned on staging.
// ---------------------------------------------------------------------------
__global__ __launch_bounds__(256)
void k_gemm_bf16(const unsigned short* __restrict__ A,
                 const unsigned short* __restrict__ Bw,
                 float* __restrict__ C, int M, int N, int K) {
  constexpr int TM = 128, TN = 256, TK = 32, RS = 40;  // RS: padded row stride
  __shared__ unsigned short As[2][TM * RS];
  __shared__ unsigned short Bs[2][TN * RS];

  const int tid  = threadIdx.x;
  const int lane = tid & 31;
  const int w    = tid >> 5;
  const int wm   = w >> 2;       // 0..1
  const int wn   = w & 3;        // 0..3
  const int bm0  = blockIdx.y * TM;
  const int bn0  = blockIdx.x * TN;
  const int nk   = K / TK;

  v8f acc[4][4];
  const v8f vz = {0.f,0.f,0.f,0.f,0.f,0.f,0.f,0.f};
#pragma unroll
  for (int i = 0; i < 4; ++i)
#pragma unroll
    for (int j = 0; j < 4; ++j) acc[i][j] = vz;

#ifdef USE_TDM
  const uint32_t ldsA0 = (uint32_t)(uintptr_t)(&As[0][0]);
  const uint32_t ldsA1 = (uint32_t)(uintptr_t)(&As[1][0]);
  const uint32_t ldsB0 = (uint32_t)(uintptr_t)(&Bs[0][0]);
  const uint32_t ldsB1 = (uint32_t)(uintptr_t)(&Bs[1][0]);
  const uint32_t bh = (uint32_t)((N - bn0) < TN ? (N - bn0) : TN);
  auto issueStage = [&](int kt, int buf) {
    tdm_load_tile_bf16(A + (size_t)bm0 * K + kt * TK, buf ? ldsA1 : ldsA0,
                       TM, (uint32_t)K, TK, TM);
    tdm_load_tile_bf16(Bw + (size_t)bn0 * K + kt * TK, buf ? ldsB1 : ldsB0,
                       bh, (uint32_t)K, TK, TN);
  };
#else
  auto fetchA = [&](int kt, uint4* pa) {
#pragma unroll
    for (int j = 0; j < 2; ++j) {
      int ch = tid + j * 256;
      int row = ch >> 2, kc = (ch & 3) * 8;
      pa[j] = *(const uint4*)(A + (size_t)(bm0 + row) * K + kt * TK + kc);
    }
  };
  auto fetchB = [&](int kt, uint4* pb) {
#pragma unroll
    for (int j = 0; j < 4; ++j) {
      int ch = tid + j * 256;
      int row = ch >> 2, kc = (ch & 3) * 8;
      int n = bn0 + row;
      if (n < N)
        pb[j] = *(const uint4*)(Bw + (size_t)n * K + kt * TK + kc);
      else
        pb[j] = make_uint4(0u, 0u, 0u, 0u);
    }
  };
  auto storeA = [&](const uint4* pa, int buf) {
#pragma unroll
    for (int j = 0; j < 2; ++j) {
      int ch = tid + j * 256;
      int row = ch >> 2, kc = (ch & 3) * 8;
      *(uint4*)(&As[buf][row * RS + kc]) = pa[j];
    }
  };
  auto storeB = [&](const uint4* pb, int buf) {
#pragma unroll
    for (int j = 0; j < 4; ++j) {
      int ch = tid + j * 256;
      int row = ch >> 2, kc = (ch & 3) * 8;
      *(uint4*)(&Bs[buf][row * RS + kc]) = pb[j];
    }
  };
#endif

  // ---- fragment loads (wave32 layouts, ISA 7.12.2) ------------------------
  union FragU { struct { uint4 lo, hi; } q; v16bf v; };
  // A 16x32 bf16: lanes 0-15 -> rows 0-15 holding K {0..7, 16..23};
  //               lanes 16-31 -> rows 0-15 holding K {8..15, 24..31}
  auto loadFragA = [&](const unsigned short* base, int row0) -> v16bf {
    int r  = row0 + (lane & 15);
    int kb = (lane >> 4) << 3;             // 0 or 8
    FragU u;
    u.q.lo = *(const uint4*)(base + r * RS + kb);
    u.q.hi = *(const uint4*)(base + r * RS + kb + 16);
    return u.v;
  };
  // B 32x16 bf16 (stored as rows of W = columns of B):
  //   lanes 0-15 -> cols 0-15 holding K {0..7, 16..23}; lanes 16-31 -> K {8..15, 24..31}
  auto loadFragB = [&](const unsigned short* base, int col0) -> v16bf {
    int r  = col0 + (lane & 15);
    int kb = (lane >> 4) << 3;             // 0 or 8
    FragU u;
    u.q.lo = *(const uint4*)(base + r * RS + kb);
    u.q.hi = *(const uint4*)(base + r * RS + kb + 16);
    return u.v;
  };

  auto compute = [&](int buf) {
    v16bf bfr[4];
#pragma unroll
    for (int nt = 0; nt < 4; ++nt)
      bfr[nt] = loadFragB(&Bs[buf][0], wn * 64 + nt * 16);
#pragma unroll
    for (int mt = 0; mt < 4; ++mt) {
      v16bf afr = loadFragA(&As[buf][0], wm * 64 + mt * 16);
#pragma unroll
      for (int nt = 0; nt < 4; ++nt)
        acc[mt][nt] = __builtin_amdgcn_wmma_f32_16x16x32_bf16(
            false, afr, false, bfr[nt], (short)0, acc[mt][nt], false, false);
    }
  };

  // ---- pipelined main loop ------------------------------------------------
#ifdef USE_TDM
  if (w == 0) issueStage(0, 0);
  for (int kt = 0; kt < nk; ++kt) {
    const int cur = kt & 1;
    const bool hasNext = (kt + 1) < nk;
    if (w == 0) {
      if (hasNext) {
        issueStage(kt + 1, cur ^ 1);
        WAIT_TENSORCNT(2);   // stage kt's 2 loads complete (in-order)
      } else {
        WAIT_TENSORCNT(0);   // drain
      }
    }
    __syncthreads();         // stage kt visible to all waves
    compute(cur);
    __syncthreads();         // buf cur free for stage kt+2
  }
#else
  uint4 pa[2], pb[4];
  fetchA(0, pa); fetchB(0, pb);
  storeA(pa, 0); storeB(pb, 0);
  __syncthreads();
  for (int kt = 0; kt < nk; ++kt) {
    const int cur = kt & 1;
    const bool hasNext = (kt + 1) < nk;
    if (hasNext) { fetchA(kt + 1, pa); fetchB(kt + 1, pb); }
    compute(cur);
    __syncthreads();
    if (hasNext) { storeA(pa, cur ^ 1); storeB(pb, cur ^ 1); }
    __syncthreads();
  }
#endif

  // ---- epilogue: C/D 16x16 f32 layout (VGPR i: M=i / M=i+8 by lane half) --
#pragma unroll
  for (int mt = 0; mt < 4; ++mt) {
    int rbase = bm0 + wm * 64 + mt * 16 + ((lane >> 4) << 3);
#pragma unroll
    for (int nt = 0; nt < 4; ++nt) {
      int col = bn0 + wn * 64 + nt * 16 + (lane & 15);
      if (col < N) {
#pragma unroll
        for (int i = 0; i < 8; ++i)
          C[(size_t)(rbase + i) * N + col] = acc[mt][nt][i];
      }
    }
  }
}

// ---------------------------------------------------------------------------
// Depthwise causal conv1d (K=4) + bias + SiLU over the hbc slice of proj.
// Splits output into xs (bf16), B (f32), C (f32).
// ---------------------------------------------------------------------------
__global__ void k_conv_silu(const float* __restrict__ proj,
                            const float* __restrict__ conv_w,
                            const float* __restrict__ conv_b,
                            unsigned short* __restrict__ xs,
                            float* __restrict__ Bbuf,
                            float* __restrict__ Cbuf) {
  int idx = blockIdx.x * blockDim.x + threadIdx.x;
  const int total = B_ * L_ * CONV_DIM_;
  if (idx >= total) return;
  int c  = idx % CONV_DIM_;
  int bl = idx / CONV_DIM_;
  int l  = bl % L_;
  int b  = bl / L_;

  float acc = conv_b[c];
#pragma unroll
  for (int j = 0; j < K_; ++j) {
    int ll = l + j - (K_ - 1);
    if (ll >= 0)
      acc += proj[(size_t)(b * L_ + ll) * PROJ_ + I_ + c] * conv_w[c * K_ + j];
  }
  acc = siluf(acc);
  size_t tok = (size_t)(b * L_ + l);
  if (c < I_)               xs  [tok * I_ + c]             = f2bf(acc);
  else if (c < I_ + N_)     Bbuf[tok * N_ + (c - I_)]      = acc;
  else                      Cbuf[tok * N_ + (c - I_ - N_)] = acc;
}

// ---------------------------------------------------------------------------
// dt = softplus(dt_raw + dt_bias),  dA = exp(dt * -exp(A_log))
// ---------------------------------------------------------------------------
__global__ void k_dt(const float* __restrict__ proj,
                     const float* __restrict__ dt_bias,
                     const float* __restrict__ A_log,
                     float* __restrict__ dt_o, float* __restrict__ dA_o) {
  int i = blockIdx.x * blockDim.x + threadIdx.x;
  if (i >= B_ * L_ * H_) return;
  int h  = i % H_;
  int bl = i / H_;
  float x = proj[(size_t)bl * PROJ_ + I_ + CONV_DIM_ + h] + dt_bias[h];
  float dt = (x > 20.f) ? x : log1pf(__expf(x));
  float dA = __expf(dt * (-__expf(A_log[h])));
  dt_o[i] = dt;
  dA_o[i] = dA;
}

// ---------------------------------------------------------------------------
// Chunked SSM scan (linear recurrence decomposition).
// Thread layout (passes 1 and 3): 256 threads, thread t owns p = t>>2 and
// n-range (t&3)*16..+15 of the per-head state h[P=64][N=64] (16 regs).
// ---------------------------------------------------------------------------

// Pass 1: per-chunk local state (from zero) + per-chunk decay product.
__global__ __launch_bounds__(256)
void k_scan_local(const unsigned short* __restrict__ xs,
                  const float* __restrict__ Bbuf,
                  const float* __restrict__ dt, const float* __restrict__ dA,
                  float* __restrict__ Sloc, float* __restrict__ prodbuf) {
  const int c  = blockIdx.x % NC_;
  const int h  = (blockIdx.x / NC_) % H_;
  const int b  = blockIdx.x / (NC_ * H_);
  const int tid = threadIdx.x;
  const int p   = tid >> 2;
  const int sub = tid & 3;
  const size_t tok0 = (size_t)b * L_ + c * QC_;

  float st[16];
#pragma unroll
  for (int i = 0; i < 16; ++i) st[i] = 0.f;
  float prod = 1.f;

  __shared__ float sX[CH_][P_];
  __shared__ float sB[CH_][N_];
  __shared__ float sS[CH_][2];

  for (int t0 = 0; t0 < QC_; t0 += CH_) {
    for (int j = tid; j < CH_ * P_; j += 256)
      sX[j >> 6][j & 63] = bf2f(xs[(tok0 + t0 + (j >> 6)) * I_ + h * P_ + (j & 63)]);
    for (int j = tid; j < CH_ * N_; j += 256)
      sB[j >> 6][j & 63] = Bbuf[(tok0 + t0 + (j >> 6)) * N_ + (j & 63)];
    if (tid < CH_ * 2) {
      int tt = tid >> 1;
      size_t o = (tok0 + t0 + tt) * H_ + h;
      sS[tt][tid & 1] = (tid & 1) ? dA[o] : dt[o];
    }
    __syncthreads();
#pragma unroll
    for (int q = 0; q < CH_; ++q) {
      float dtv = sS[q][0], dAv = sS[q][1];
      float dtx = dtv * sX[q][p];
      prod *= dAv;
#pragma unroll
      for (int i = 0; i < 16; ++i) {
        int n = sub * 16 + i;
        st[i] = st[i] * dAv + dtx * sB[q][n];
      }
    }
    __syncthreads();
  }

  size_t base = ((size_t)((b * H_ + h) * NC_ + c)) * (P_ * N_);
#pragma unroll
  for (int i = 0; i < 16; ++i)
    Sloc[base + p * N_ + sub * 16 + i] = st[i];
  if (tid == 0) prodbuf[(b * H_ + h) * NC_ + c] = prod;
}

// Pass 2: exclusive prefix over chunks: Sini[c] = carry; carry = carry*prod[c] + Sloc[c]
__global__ __launch_bounds__(256)
void k_scan_prefix(const float* __restrict__ Sloc,
                   const float* __restrict__ prodbuf,
                   float* __restrict__ Sini) {
  const int bh  = blockIdx.x;   // b*H + h
  const int tid = threadIdx.x;
  float carry[16];
#pragma unroll
  for (int k = 0; k < 16; ++k) carry[k] = 0.f;

  for (int c = 0; c < NC_; ++c) {
    size_t base = ((size_t)(bh * NC_ + c)) * (P_ * N_);
    float pr = prodbuf[bh * NC_ + c];
#pragma unroll
    for (int k = 0; k < 16; ++k) {
      int idx = tid + k * 256;
      float sl = Sloc[base + idx];
      Sini[base + idx] = carry[k];
      carry[k] = carry[k] * pr + sl;
    }
  }
}

// Pass 3: rerun each chunk from its corrected initial state, emitting y.
__global__ __launch_bounds__(256)
void k_scan_apply(const unsigned short* __restrict__ xs,
                  const float* __restrict__ Bbuf, const float* __restrict__ Cbuf,
                  const float* __restrict__ dt, const float* __restrict__ dA,
                  const float* __restrict__ Sini,
                  const float* __restrict__ Dv, float* __restrict__ y) {
  const int c  = blockIdx.x % NC_;
  const int h  = (blockIdx.x / NC_) % H_;
  const int b  = blockIdx.x / (NC_ * H_);
  const int tid = threadIdx.x;
  const int p   = tid >> 2;
  const int sub = tid & 3;
  const size_t tok0 = (size_t)b * L_ + c * QC_;
  const float Dh = Dv[h];

  float st[16];
  {
    size_t base = ((size_t)((b * H_ + h) * NC_ + c)) * (P_ * N_);
#pragma unroll
    for (int i = 0; i < 16; ++i)
      st[i] = Sini[base + p * N_ + sub * 16 + i];
  }

  __shared__ float sX[CH_][P_];
  __shared__ float sB[CH_][N_];
  __shared__ float sC[CH_][N_];
  __shared__ float sS[CH_][2];

  for (int t0 = 0; t0 < QC_; t0 += CH_) {
    for (int j = tid; j < CH_ * P_; j += 256)
      sX[j >> 6][j & 63] = bf2f(xs[(tok0 + t0 + (j >> 6)) * I_ + h * P_ + (j & 63)]);
    for (int j = tid; j < CH_ * N_; j += 256)
      sB[j >> 6][j & 63] = Bbuf[(tok0 + t0 + (j >> 6)) * N_ + (j & 63)];
    for (int j = tid; j < CH_ * N_; j += 256)
      sC[j >> 6][j & 63] = Cbuf[(tok0 + t0 + (j >> 6)) * N_ + (j & 63)];
    if (tid < CH_ * 2) {
      int tt = tid >> 1;
      size_t o = (tok0 + t0 + tt) * H_ + h;
      sS[tt][tid & 1] = (tid & 1) ? dA[o] : dt[o];
    }
    __syncthreads();
#pragma unroll
    for (int q = 0; q < CH_; ++q) {
      float dtv = sS[q][0], dAv = sS[q][1];
      float xv  = sX[q][p];
      float dtx = dtv * xv;
      float accv = 0.f;
#pragma unroll
      for (int i = 0; i < 16; ++i) {
        int n = sub * 16 + i;
        st[i] = st[i] * dAv + dtx * sB[q][n];
        accv += st[i] * sC[q][n];
      }
      accv += __shfl_xor(accv, 1);
      accv += __shfl_xor(accv, 2);
      if (sub == 0) y[(tok0 + t0 + q) * I_ + h * P_ + p] = accv + Dh * xv;
    }
    __syncthreads();
  }
}

// ---------------------------------------------------------------------------
// Gated RMSNorm: h = y * silu(gate); h *= rsqrt(mean(h^2)+eps); out = w*h (bf16)
// One workgroup per token (I=4096, 256 threads x 16 elements).
// ---------------------------------------------------------------------------
__global__ __launch_bounds__(256)
void k_gated_rmsnorm(const float* __restrict__ y,
                     const float* __restrict__ proj,   // gate = proj[:, :I]
                     const float* __restrict__ w,
                     unsigned short* __restrict__ out) {
  const int tok = blockIdx.x;
  const int tid = threadIdx.x;
  const size_t ybase = (size_t)tok * I_;
  const size_t gbase = (size_t)tok * PROJ_;

  float vals[16];
  float ss = 0.f;
#pragma unroll
  for (int i = 0; i < 16; ++i) {
    int idx = tid + i * 256;
    float g  = proj[gbase + idx];
    float hv = y[ybase + idx] * siluf(g);
    vals[i] = hv;
    ss += hv * hv;
  }
#pragma unroll
  for (int o = 16; o > 0; o >>= 1) ss += __shfl_xor(ss, o);
  __shared__ float sred[8];
  if ((tid & 31) == 0) sred[tid >> 5] = ss;
  __syncthreads();
  __shared__ float sscale;
  if (tid == 0) {
    float t = 0.f;
#pragma unroll
    for (int i = 0; i < 8; ++i) t += sred[i];
    sscale = rsqrtf(t / (float)I_ + EPS_);
  }
  __syncthreads();
  const float sc = sscale;
#pragma unroll
  for (int i = 0; i < 16; ++i) {
    int idx = tid + i * 256;
    out[ybase + idx] = f2bf(vals[i] * sc * w[idx]);
  }
}

// ---------------------------------------------------------------------------
// host-side orchestration
// ---------------------------------------------------------------------------
extern "C" void kernel_launch(void* const* d_in, const int* in_sizes, int n_in,
                              void* d_out, int out_size, void* d_ws,
                              size_t ws_size, hipStream_t stream) {
  const float* hs      = (const float*)d_in[0];  // [B,L,DM]
  const float* w_in    = (const float*)d_in[1];  // [PROJ,DM]
  const float* conv_w  = (const float*)d_in[2];  // [CONV_DIM,K]
  const float* conv_b  = (const float*)d_in[3];  // [CONV_DIM]
  const float* dt_bias = (const float*)d_in[4];  // [H]
  const float* A_log   = (const float*)d_in[5];  // [H]
  const float* Dv      = (const float*)d_in[6];  // [H]
  const float* norm_w  = (const float*)d_in[7];  // [I]
  const float* w_out   = (const float*)d_in[8];  // [DM,I]
  float* out = (float*)d_out;

  char* p = (char*)d_ws;
  auto alloc = [&](size_t bytes) -> void* {
    void* r = (void*)p;
    p += (bytes + 255) & ~(size_t)255;
    return r;
  };
  unsigned short* w1b  = (unsigned short*)alloc((size_t)PROJ_ * DM_ * 2);
  unsigned short* w2b  = (unsigned short*)alloc((size_t)DM_ * I_ * 2);
  unsigned short* xb   = (unsigned short*)alloc((size_t)B_ * L_ * DM_ * 2);
  float*          proj = (float*)alloc((size_t)B_ * L_ * PROJ_ * 4);
  unsigned short* xs   = (unsigned short*)alloc((size_t)B_ * L_ * I_ * 2);
  float*          Bbuf = (float*)alloc((size_t)B_ * L_ * N_ * 4);
  float*          Cbuf = (float*)alloc((size_t)B_ * L_ * N_ * 4);
  float*          dtb  = (float*)alloc((size_t)B_ * L_ * H_ * 4);
  float*          dAb  = (float*)alloc((size_t)B_ * L_ * H_ * 4);
  float*          yb   = (float*)alloc((size_t)B_ * L_ * I_ * 4);
  unsigned short* ynb  = (unsigned short*)alloc((size_t)B_ * L_ * I_ * 2);
  float*          Sloc = (float*)alloc((size_t)B_ * H_ * NC_ * P_ * N_ * 4);
  float*          Sini = (float*)alloc((size_t)B_ * H_ * NC_ * P_ * N_ * 4);
  float*          prodb= (float*)alloc((size_t)B_ * H_ * NC_ * 4);

  const int M = B_ * L_;  // 4096 tokens

  // 1) fp32 -> bf16 conversions
  {
    int n1 = PROJ_ * DM_;
    k_cvt_bf16<<<(n1 + 255) / 256, 256, 0, stream>>>(w_in, w1b, n1);
    int n2 = DM_ * I_;
    k_cvt_bf16<<<(n2 + 255) / 256, 256, 0, stream>>>(w_out, w2b, n2);
    int n3 = M * DM_;
    k_cvt_bf16<<<(n3 + 255) / 256, 256, 0, stream>>>(hs, xb, n3);
  }

  // 2) in_proj GEMM: proj[M, PROJ] = xb[M, DM] @ w1b[PROJ, DM]^T
  {
    dim3 grid((PROJ_ + 255) / 256, M / 128);
    k_gemm_bf16<<<grid, 256, 0, stream>>>(xb, w1b, proj, M, PROJ_, DM_);
  }

  // 3) depthwise causal conv + SiLU, split outputs
  {
    int total = B_ * L_ * CONV_DIM_;
    k_conv_silu<<<(total + 255) / 256, 256, 0, stream>>>(proj, conv_w, conv_b,
                                                         xs, Bbuf, Cbuf);
  }

  // 4) dt softplus + dA
  {
    int total = B_ * L_ * H_;
    k_dt<<<(total + 255) / 256, 256, 0, stream>>>(proj, dt_bias, A_log, dtb,
                                                  dAb);
  }

  // 5) chunked SSM scan: local states -> prefix -> apply
  k_scan_local<<<B_ * H_ * NC_, 256, 0, stream>>>(xs, Bbuf, dtb, dAb, Sloc,
                                                  prodb);
  k_scan_prefix<<<B_ * H_, 256, 0, stream>>>(Sloc, prodb, Sini);
  k_scan_apply<<<B_ * H_ * NC_, 256, 0, stream>>>(xs, Bbuf, Cbuf, dtb, dAb,
                                                  Sini, Dv, yb);

  // 6) gated RMSNorm -> bf16
  k_gated_rmsnorm<<<M, 256, 0, stream>>>(yb, proj, norm_w, ynb);

  // 7) out_proj GEMM: out[M, DM] = ynb[M, I] @ w2b[DM, I]^T
  {
    dim3 grid(DM_ / 256, M / 128);
    k_gemm_bf16<<<grid, 256, 0, stream>>>(ynb, w2b, out, M, DM_, I_);
  }
}
